// SCS_47725676593200
// MI455X (gfx1250) — matrix operand order repaired and checked by reading, compiled, not run on hardware
//
#include <hip/hip_runtime.h>
#include <hip/hip_bf16.h>
#include <math.h>

// ---------------------------------------------------------------------------
// SCS (sharpened cosine similarity) conv as implicit GEMM on CDNA5 WMMA.
//   x: [16,64,128,128] f32   w: [576,128] f32 (rows ordered kh,kw,c)
//   out: [16,128,128,128] f32
// GEMM: M=262144 pixels, N=128, K=576.  BF16 WMMA 16x16x32, f32 accum.
// One workgroup (256 thr = 8 wave32) per (b,h) row: 128 pixels x 128 chans.
// Weights pre-packed to bf16 LDS image in d_ws; per K-step a 10KB block is
// DMA'd into LDS by the Tensor Data Mover (double-buffered on TENSORcnt).
// ---------------------------------------------------------------------------

typedef __attribute__((ext_vector_type(16))) __bf16 v16bf;
typedef __attribute__((ext_vector_type(8)))  __bf16 v8bf;
typedef __attribute__((ext_vector_type(8)))  float  v8f;
typedef __attribute__((ext_vector_type(4)))  unsigned int u32x4;
typedef __attribute__((ext_vector_type(8)))  int i32x8;
typedef __attribute__((ext_vector_type(4)))  int i32x4;

#define BATCH 16
#define CIN   64
#define HH    128
#define WW    128
#define COUT  128
#define KTOT  576              // 3*3*64, patch order (kh,kw,c)
#define KSTEP 32
#define NKS   (KTOT / KSTEP)   // 18 K-steps; each lies inside one (kh,kw) cell
#define LDA   40               // padded LDS row stride in bf16 elems (80B)
#define BBLK  (128 * LDA * 2)  // one staged tile: 10240 bytes (2560 dwords)
#define WPACK_OFF 256          // wpack offset inside d_ws (scalars live below)

// --------------------------- prep: scalars ---------------------------------
__global__ __launch_bounds__(256)
void scs_prep(const float* __restrict__ w, const float* __restrict__ q,
              const float* __restrict__ p, float* __restrict__ scal) {
  __shared__ float red[256];
  const int tid = threadIdx.x;
  float acc = 0.f;
  for (int i = tid; i < KTOT * COUT; i += 256) { float v = w[i]; acc += v * v; }
  red[tid] = acc;
  __syncthreads();
  for (int s = 128; s > 0; s >>= 1) {
    if (tid < s) red[tid] += red[tid + s];
    __syncthreads();
  }
  if (tid == 0) {
    scal[0] = sqrtf(red[0]);  // ||w||_F
    scal[1] = expf(q[0]);     // qe
    scal[2] = expf(p[0]);     // pe
  }
}

// ---- prep: pack weights as the bf16 LDS image [ks][n][LDA] (TDM source) ----
__global__ __launch_bounds__(256)
void scs_pack(const float* __restrict__ w, __bf16* __restrict__ wpack) {
  const int e = blockIdx.x * 256 + threadIdx.x;   // 18*32*128 = 73728 total
  if (e >= NKS * KSTEP * COUT) return;
  const int n  = e & 127;
  const int kl = (e >> 7) & 31;
  const int ks = e >> 12;
  wpack[ks * (128 * LDA) + n * LDA + kl] =
      (__bf16)w[(size_t)(ks * KSTEP + kl) * COUT + n];
}

// ---------- TDM: 1-D copy of one 2560-dword weight block into LDS ----------
__device__ __forceinline__ void tdm_copy_block(const __bf16* src, unsigned lds_byte) {
  const unsigned long long ga = (unsigned long long)(uintptr_t)src;
  // D# group0: count=1 | lds_addr | global_addr[56:0] | type=2
  u32x4 g0 = { 1u, lds_byte, (unsigned)ga,
               (unsigned)(ga >> 32) | 0x80000000u };
  // D# group1: data_size=4B; tensor_dim0 = tile_dim0 = 2560 dwords (1-D)
  i32x8 g1 = { 0x00020000, (int)0x0A000000u, 0, (int)0x0A000000u, 0, 0, 0, 0 };
  i32x4 z4 = { 0, 0, 0, 0 };
#if defined(__clang_major__) && __clang_major__ >= 23
  i32x8 z8 = { 0, 0, 0, 0, 0, 0, 0, 0 };
  __builtin_amdgcn_tensor_load_to_lds(g0, g1, z4, z4, z8, 0);
#else
  __builtin_amdgcn_tensor_load_to_lds(g0, g1, z4, z4, 0);
#endif
}

// --------------------------- main kernel -----------------------------------
template <bool USE_TDM>
__global__ __launch_bounds__(256)
void scs_main(const float* __restrict__ x, const float* __restrict__ w,
              const float* __restrict__ bias, const float* __restrict__ scal,
              const __bf16* __restrict__ wpack, float* __restrict__ out) {
  // [0,10240):        A tile (128 px x 32 k bf16, padded rows)
  // [10240,20480):    B tile buffer 0   (TDM destination / staging)
  // [20480,30720):    B tile buffer 1   (TDM double buffer)
  // Epilogue aliases [0,8704) as per-wave f32 transpose scratch.
  __shared__ __align__(16) unsigned char s_pool[3 * BBLK];   // 30 KB
  __shared__ float s_ps[128];                                // ||s||^2 per px
  __bf16* At = (__bf16*)s_pool;

  const int tid  = threadIdx.x;
  const int bb   = blockIdx.x >> 7;   // batch
  const int h    = blockIdx.x & 127;  // output row

  const int lane = tid & 31;
  const int wv   = tid >> 5;          // wave id 0..7 -> N chunk [16*wv, +16)
  const int hi   = lane >> 4;         // fragment half selector
  const int ln   = lane & 15;

  const int p    = tid & 127;         // loader: pixel (w coordinate)
  const int klo  = tid >> 7;          // loader: k parity (0/1)

  v8f acc[8] = {};                    // 8 x (16n x 16m) f32 accumulators
  float sumsq = 0.f;                  // this thread's share of patch ||s||^2

  const unsigned lds_b0 = (unsigned)(uintptr_t)(void*)(s_pool + BBLK);

  // Pre-issue TDM for K-step 0 into buffer 0.
  if (USE_TDM && wv == 0) tdm_copy_block(wpack, lds_b0);

  for (int ks = 0; ks < NKS; ++ks) {
    const int k0    = ks * KSTEP;
    const int cell  = k0 >> 6;        // kh*3 + kw
    const int cbase = k0 & 32;        // channel base within 64-ch cell
    const int dh    = cell / 3 - 1;
    const int dw    = cell % 3 - 1;
    const int hs    = h + dh;
    const bool hok  = (unsigned)hs < (unsigned)HH;
    const int  wsrc = p + dw;
    const bool wok  = (unsigned)wsrc < (unsigned)WW;
    const bool ok   = hok && wok;

    __syncthreads();   // previous step's fragment reads complete

    // ---- TDM: prefetch next weight block into the other buffer ----
    if (USE_TDM && wv == 0 && ks + 1 < NKS)
      tdm_copy_block(wpack + (size_t)(ks + 1) * (128 * LDA),
                     lds_b0 + (unsigned)(((ks + 1) & 1) * BBLK));

    // ---- stage A: x[bb, cbase..+31, hs, p+dw] -> At[p][k], f32 -> bf16.
    //      Channel pairs -> one merged 4B LDS store per pair.
    {
      const size_t baseA =
          (((size_t)bb * CIN + cbase) * HH + (hok ? hs : 0)) * (size_t)WW +
          (wok ? wsrc : 0);
#pragma unroll
      for (int j = 0; j < 8; ++j) {
        const int kl0 = 4 * j + 2 * klo;
        float v0 = 0.f, v1 = 0.f;
        if (ok) {
          v0 = x[baseA + (size_t)kl0 * (HH * WW)];
          v1 = x[baseA + (size_t)(kl0 + 1) * (HH * WW)];
        }
        sumsq += v0 * v0 + v1 * v1;           // patch norm for free
        union { __bf16 h2[2]; unsigned u; } pk;
        pk.h2[0] = (__bf16)v0;
        pk.h2[1] = (__bf16)v1;
        *(unsigned*)&At[p * LDA + kl0] = pk.u;
      }
    }

    // ---- fallback B staging (no TDM): w f32 -> Bt0 bf16, transposed ----
    if (!USE_TDM) {
      __bf16* Bt = (__bf16*)(s_pool + BBLK);
      const float* wb = w + (size_t)(k0 + klo) * COUT + p;  // p plays n role
#pragma unroll
      for (int j = 0; j < 16; ++j)
        Bt[p * LDA + (2 * j + klo)] = (__bf16)wb[(size_t)(2 * j) * COUT];
    }

    // ---- prefetch next K-step's x row into L2 (global_prefetch_b8) ----
    if (ks + 1 < NKS) {
      const int k0n = k0 + KSTEP;
      const int cn  = k0n >> 6;
      const int hn  = h + cn / 3 - 1;
      const int wn  = p + cn % 3 - 1;
      if ((unsigned)hn < (unsigned)HH && (unsigned)wn < (unsigned)WW)
        __builtin_prefetch(
            x + (((size_t)bb * CIN + (k0n & 32) + klo) * HH + hn) * (size_t)WW + wn,
            0, 1);
    }

    // TENSORcnt is in-order per wave: <=1 outstanding means this step's
    // block has landed (the newer one may still be in flight).
    if (USE_TDM && wv == 0) {
      if (ks + 1 < NKS) __builtin_amdgcn_s_wait_tensorcnt(1);
      else              __builtin_amdgcn_s_wait_tensorcnt(0);
    }

    __syncthreads();   // tiles visible

    // ---- MMA: each wave: 1 B frag (its 16 N) x 8 A frags (128 pixels).
    //      bf16 fragment layout per ISA 7.12.2: two contiguous 16B LDS reads.
    const __bf16* Bt =
        (const __bf16*)(s_pool + BBLK + (USE_TDM ? (ks & 1) * BBLK : 0));
    const int n = wv * 16 + ln;
    union { v16bf v; v8bf h2[2]; } bfr, afr;
    bfr.h2[0] = *(const v8bf*)&Bt[n * LDA + hi * 8];
    bfr.h2[1] = *(const v8bf*)&Bt[n * LDA + 16 + hi * 8];
#pragma unroll
    for (int mb = 0; mb < 8; ++mb) {
      const int m = mb * 16 + ln;
      afr.h2[0] = *(const v8bf*)&At[m * LDA + hi * 8];
      afr.h2[1] = *(const v8bf*)&At[m * LDA + 16 + hi * 8];
      acc[mb] = __builtin_amdgcn_wmma_f32_16x16x32_bf16(
          false, afr.v, false, bfr.v, (short)0, acc[mb], false, false);
    }
  }

  // ---- combine per-pixel patch norms (threads t and t+128 share pixel t&127)
  __syncthreads();
  if (tid < 128) s_ps[tid] = sumsq;
  __syncthreads();
  if (tid >= 128) s_ps[tid - 128] += sumsq;
  __syncthreads();

  const float wnorm = scal[0];
  const float qe    = scal[1];
  const float pe    = scal[2];
  const float nk    = wnorm + qe;

  // ---- epilogue: transpose each 16x16 accum through per-wave LDS scratch so
  //      global stores are contiguous along W, then apply SCS math.
  //      FULLY unrolled: every acc[mb] index must be a constant so the
  //      accumulator array stays in VGPRs (unroll-by-1 here caused the whole
  //      array to be demoted to scratch, spilling around every WMMA).
  float* obuf = (float*)s_pool + wv * (16 * 17);   // aliases dead A staging

#pragma unroll
  for (int mb = 0; mb < 8; ++mb) {
    // accum (vgpr i, lane): row m = mb*16 + i + 8*hi (pixel), col n = ln
#pragma unroll
    for (int i = 0; i < 8; ++i)
      obuf[ln * 17 + (i + hi * 8)] = acc[mb][i];
    asm volatile("s_wait_dscnt 0" ::: "memory");   // same-wave LDS RAW fence
#pragma unroll
    for (int r = 0; r < 8; ++r) {
      const int nl  = r * 2 + hi;
      const int n   = wv * 16 + nl;
      const int pp  = mb * 16 + ln;                // lanes 0..15 -> contig w
      const float dot = obuf[nl * 17 + ln];
      const float ns  = sqrtf(s_ps[pp]) + qe;
      const float sim = dot / (ns * nk);
      // |sim|^pe via hardware v_log_f32/v_exp_f32 (|sim|=0 -> exp(-inf)=0)
      const float mag = __expf(pe * __logf(fabsf(sim)));
      const float val = copysignf(mag, dot) + bias[n];
      out[(((size_t)bb * COUT + n) * HH + h) * (size_t)WW + pp] = val;
    }
    asm volatile("s_wait_dscnt 0" ::: "memory");   // WAR fence before reuse
  }
}

// --------------------------- launcher --------------------------------------
extern "C" void kernel_launch(void* const* d_in, const int* in_sizes, int n_in,
                              void* d_out, int out_size, void* d_ws, size_t ws_size,
                              hipStream_t stream) {
  (void)in_sizes; (void)n_in; (void)out_size;
  const float* x  = (const float*)d_in[0];
  const float* w  = (const float*)d_in[1];
  const float* b  = (const float*)d_in[2];
  const float* q  = (const float*)d_in[3];
  const float* p  = (const float*)d_in[4];
  float* out  = (float*)d_out;
  float* scal = (float*)d_ws;   // [0]=||w||_F, [1]=e^q, [2]=e^p

  scs_prep<<<1, 256, 0, stream>>>(w, q, p, scal);

  const size_t need = (size_t)WPACK_OFF + (size_t)NKS * BBLK;
  if (ws_size >= need) {
    __bf16* wpack = (__bf16*)((char*)d_ws + WPACK_OFF);
    scs_pack<<<(NKS * KSTEP * COUT + 255) / 256, 256, 0, stream>>>(w, wpack);
    scs_main<true><<<BATCH * HH, 256, 0, stream>>>(x, w, b, scal, wpack, out);
  } else {
    scs_main<false><<<BATCH * HH, 256, 0, stream>>>(x, w, b, scal, nullptr, out);
  }
}
